// nodeBlock_4449586119104
// MI455X (gfx1250) — matrix operand order repaired and checked by reading, not compiled, not run on hardware
//
#include <hip/hip_runtime.h>
#include <hip/hip_bf16.h>

// ---------------------------------------------------------------------------
// Fused nodeBlock: concat -> MLP(320->128->128->128, GELU,GELU,linear) -> LN
//                  -> segment-sum scatter.  bf16 WMMA (16x16x32), f32 accum.
// ---------------------------------------------------------------------------

typedef __attribute__((ext_vector_type(16))) __bf16 v16bf;
typedef __attribute__((ext_vector_type(8)))  float  v8f;

#define NELEM_C   50000
#define MAXN_C    8
#define NROWS_C   (NELEM_C * MAXN_C)      // 400000
#define TILES_C   (NROWS_C / 16)          // 25000
#define BLOCKS_C  (TILES_C / 8)           // 3125

union BF16x16 {
    v16bf v;
    __bf16 h[16];
    unsigned u[8];
    uint4 q[2];
};

static __device__ __forceinline__ unsigned pk2bf(float a, float b) {
    unsigned ua = __builtin_bit_cast(unsigned, a);
    unsigned ub = __builtin_bit_cast(unsigned, b);
    ua = (ua + 0x7FFFu + ((ua >> 16) & 1u)) >> 16;
    ub = (ub + 0x7FFFu + ((ub >> 16) & 1u)) >> 16;
    return (ua & 0xFFFFu) | (ub << 16);
}

static __device__ __forceinline__ float gelu_exact(float x) {
    return 0.5f * x * (1.0f + erff(x * 0.70710678118654752f));
}

// Build a 16x32 bf16 A-fragment for this lane from a row of f32 features.
// Per ISA layout: e0..7 = k{kb..kb+7}, e8..15 = k{kb+16..kb+23}; p already
// points at (row, kb).  Uses HW packed f32->bf16 conversion (RNE).
static __device__ __forceinline__ BF16x16 buildA(const float* __restrict__ p) {
    float f[16];
    *(float4*)(f + 0)  = *(const float4*)(p + 0);
    *(float4*)(f + 4)  = *(const float4*)(p + 4);
    *(float4*)(f + 8)  = *(const float4*)(p + 16);
    *(float4*)(f + 12) = *(const float4*)(p + 20);
    BF16x16 a;
    #pragma unroll
    for (int i = 0; i < 16; ++i) a.h[i] = (__bf16)f[i];
    return a;
}

// 8 n-tiles of one k-chunk: batch-load all B fragments, then 8 WMMAs.
template <int NPAIR>   // bf16-pairs per n-row in LDS weight array (160 or 64)
static __device__ __forceinline__ void wmma8(const unsigned* shW,
                                             int lm, int half, int kc,
                                             const BF16x16& a, v8f acc[8]) {
    BF16x16 bf[8];
    #pragma unroll
    for (int nt = 0; nt < 8; ++nt) {
        const int n = nt * 16 + lm;
        const uint4* q = (const uint4*)&shW[n * NPAIR + kc * 16 + half * 4];
        bf[nt].q[0] = q[0];
        bf[nt].q[1] = q[2];
    }
    #pragma unroll
    for (int nt = 0; nt < 8; ++nt) {
        acc[nt] = __builtin_amdgcn_wmma_f32_16x16x32_bf16(
            false, a.v, false, bf[nt].v, (short)0, acc[nt], false, false);
    }
}

// 16x128 @ 128x128 with A sourced from bf16 LDS scratch.
static __device__ __forceinline__ void gemm128(const unsigned short* hptr,
                                               const unsigned* shW,
                                               int lm, int half, v8f acc[8]) {
    #pragma unroll
    for (int nt = 0; nt < 8; ++nt) acc[nt] = (v8f){0,0,0,0,0,0,0,0};
    #pragma unroll
    for (int kc = 0; kc < 4; ++kc) {
        const int kb = kc * 32 + half * 8;
        BF16x16 a;
        a.q[0] = *(const uint4*)&hptr[lm * 128 + kb];
        a.q[1] = *(const uint4*)&hptr[lm * 128 + kb + 16];
        wmma8<64>(shW, lm, half, kc, a, acc);
    }
}

__global__ __launch_bounds__(256, 1)
void fused_mlp_ln_scatter(const float* __restrict__ gN,   // [NROWS][128]
                          const float* __restrict__ eN,   // [NROWS][128]
                          const float* __restrict__ el,   // [NELEM][64]
                          const float* __restrict__ W1,   // [320][128]
                          const float* __restrict__ b1,
                          const float* __restrict__ W2,   // [128][128]
                          const float* __restrict__ b2,
                          const float* __restrict__ W3,   // [128][128]
                          const float* __restrict__ b3,
                          const float* __restrict__ gamma,
                          const float* __restrict__ beta,
                          const int*   __restrict__ idx,  // [NROWS]
                          float* __restrict__ out)        // [DIMSIZE][128]
{
    // ---- LDS: transposed bf16 weights [n][k] (pair-packed), biases, scratch
    __shared__ unsigned       shW1[128 * 160];   // 80 KB
    __shared__ unsigned       shW2[128 * 64];    // 32 KB
    __shared__ unsigned       shW3[128 * 64];    // 32 KB
    __shared__ float          shBias[5 * 128];   // b1,b2,b3,gamma,beta
    __shared__ unsigned short shH[8][16 * 128];  // per-wave activations, 32 KB

    const int tid = threadIdx.x;

    // stage W^T as bf16 pairs: shW[n*NPAIR + k/2]
    for (int p = tid; p < 128 * 160; p += 256) {
        int n  = p / 160;
        int kp = (p - n * 160) * 2;
        shW1[p] = pk2bf(W1[kp * 128 + n], W1[(kp + 1) * 128 + n]);
    }
    for (int p = tid; p < 128 * 64; p += 256) {
        int n  = p / 64;
        int kp = (p - n * 64) * 2;
        shW2[p] = pk2bf(W2[kp * 128 + n], W2[(kp + 1) * 128 + n]);
        shW3[p] = pk2bf(W3[kp * 128 + n], W3[(kp + 1) * 128 + n]);
    }
    if (tid < 128) {
        shBias[tid]       = b1[tid];
        shBias[128 + tid] = b2[tid];
        shBias[256 + tid] = b3[tid];
        shBias[384 + tid] = gamma[tid];
        shBias[512 + tid] = beta[tid];
    }
    __syncthreads();

    const int wave = tid >> 5;
    const int lane = tid & 31;
    const int half = lane >> 4;      // 0: lanes 0-15, 1: lanes 16-31
    const int lm   = lane & 15;

    const int tile = blockIdx.x * 8 + wave;   // 0 .. 24999 (exact)
    const int row0 = tile * 16;

    unsigned short* hptr = shH[wave];

    // =====================  Layer 1: x[16x320] @ W1  =====================
    v8f acc[8];
    #pragma unroll
    for (int nt = 0; nt < 8; ++nt) acc[nt] = (v8f){0,0,0,0,0,0,0,0};

    const int    rowA = row0 + lm;                     // A-matrix row for lane
    const float* srcG = gN + (size_t)rowA * 128 + half * 8;
    const float* srcE = eN + (size_t)rowA * 128 + half * 8;
    const float* srcL = el + (size_t)(rowA >> 3) * 64 + half * 8;

    #pragma unroll
    for (int kc = 0; kc < 4; ++kc)       // k = 0..127 : gatheredNodes
        wmma8<160>(shW1, lm, half, kc, buildA(srcG + kc * 32), acc);
    #pragma unroll
    for (int kc = 0; kc < 4; ++kc)       // k = 128..255 : elementNodes
        wmma8<160>(shW1, lm, half, kc + 4, buildA(srcE + kc * 32), acc);
    #pragma unroll
    for (int kc = 0; kc < 2; ++kc)       // k = 256..319 : elements (bcast)
        wmma8<160>(shW1, lm, half, kc + 8, buildA(srcL + kc * 32), acc);

    // bias + GELU -> LDS scratch (bf16, row-major [16][128])
    #pragma unroll
    for (int nt = 0; nt < 8; ++nt) {
        const int n = nt * 16 + lm;
        const float bb = shBias[n];
        #pragma unroll
        for (int v = 0; v < 8; ++v) {
            const float g = gelu_exact(acc[nt][v] + bb);
            hptr[(v + half * 8) * 128 + n] = __builtin_bit_cast(unsigned short, (__bf16)g);
        }
    }
    __syncthreads();

    // =====================  Layer 2: h1[16x128] @ W2  =====================
    gemm128(hptr, shW2, lm, half, acc);
    __syncthreads();

    #pragma unroll
    for (int nt = 0; nt < 8; ++nt) {
        const int n = nt * 16 + lm;
        const float bb = shBias[128 + n];
        #pragma unroll
        for (int v = 0; v < 8; ++v) {
            const float g = gelu_exact(acc[nt][v] + bb);
            hptr[(v + half * 8) * 128 + n] = __builtin_bit_cast(unsigned short, (__bf16)g);
        }
    }
    __syncthreads();

    // =====================  Layer 3: h2[16x128] @ W3  =====================
    gemm128(hptr, shW3, lm, half, acc);

    // ===============  LayerNorm (over N=128) + scatter-add  ===============
    const float rH = 1.0f / 128.0f;
    #pragma unroll
    for (int v = 0; v < 8; ++v) {
        float vals[8];
        float s = 0.0f, s2 = 0.0f;
        #pragma unroll
        for (int nt = 0; nt < 8; ++nt) {
            const float x = acc[nt][v] + shBias[256 + nt * 16 + lm];
            vals[nt] = x;
            s  += x;
            s2 += x * x;
        }
        // reduce across the 16-lane half that owns this row
        #pragma unroll
        for (int m = 1; m <= 8; m <<= 1) {
            s  += __shfl_xor(s,  m, 32);
            s2 += __shfl_xor(s2, m, 32);
        }
        const float mu   = s * rH;
        const float var  = s2 * rH - mu * mu;
        const float rstd = rsqrtf(var + 1e-5f);

        const int r = row0 + v + half * 8;
        const int d = idx[r];
        float* op = out + (size_t)d * 128;
        #pragma unroll
        for (int nt = 0; nt < 8; ++nt) {
            const int n = nt * 16 + lm;
            const float y = (vals[nt] - mu) * rstd * shBias[384 + n] + shBias[512 + n];
            atomicAdd(op + n, y);
        }
    }
}

extern "C" void kernel_launch(void* const* d_in, const int* in_sizes, int n_in,
                              void* d_out, int out_size, void* d_ws, size_t ws_size,
                              hipStream_t stream) {
    (void)in_sizes; (void)n_in; (void)d_ws; (void)ws_size;
    const float* gN    = (const float*)d_in[0];
    const float* eN    = (const float*)d_in[1];
    const float* el    = (const float*)d_in[2];
    const float* W1    = (const float*)d_in[3];
    const float* b1    = (const float*)d_in[4];
    const float* W2    = (const float*)d_in[5];
    const float* b2    = (const float*)d_in[6];
    const float* W3    = (const float*)d_in[7];
    const float* b3    = (const float*)d_in[8];
    const float* gamma = (const float*)d_in[9];
    const float* beta  = (const float*)d_in[10];
    const int*   idx   = (const int*)d_in[11];

    hipMemsetAsync(d_out, 0, (size_t)out_size * sizeof(float), stream);
    fused_mlp_ln_scatter<<<dim3(BLOCKS_C), dim3(256), 0, stream>>>(
        gN, eN, el, W1, b1, W2, b2, W3, b3, gamma, beta, idx, (float*)d_out);
}
